// Head_25451976196235
// MI455X (gfx1250) — 8-bit matrix operand order repaired and checked by reading, compiled, not run on hardware
//
#include <hip/hip_runtime.h>
#include <stdint.h>

typedef int v8i __attribute__((ext_vector_type(8)));

#define B_DIM 512
#define T_DIM 256
#define H_DIM 64
#define NELEM (B_DIM*T_DIM*H_DIM)

__global__ void ibert_init(unsigned* ws, float* out) {
    if (threadIdx.x == 0 && blockIdx.x == 0) {
        ws[0] = 0u; ws[1] = 0u; ws[2] = 0u;
        out[NELEM] = 0.00390625f;
    }
}

__global__ void absmax_kernel(const float* q, const float* k, const float* v, unsigned* ws) {
    const float* p = (blockIdx.y == 0) ? q : (blockIdx.y == 1) ? k : v;
    const float4* p4 = (const float4*)p;
    const int n4 = NELEM / 4;
    float m = 0.f;
    for (int i = blockIdx.x * blockDim.x + threadIdx.x; i < n4; i += gridDim.x * blockDim.x) {
        float4 t = p4[i];
        m = fmaxf(m, fmaxf(fmaxf(fabsf(t.x), fabsf(t.y)), fmaxf(fabsf(t.z), fabsf(t.w))));
    }
    __shared__ float red[256];
    red[threadIdx.x] = m;
    __syncthreads();
    for (int s = 128; s > 0; s >>= 1) {
        if (threadIdx.x < s) red[threadIdx.x] = fmaxf(red[threadIdx.x], red[threadIdx.x + s]);
        __syncthreads();
    }
    if (threadIdx.x == 0) atomicMax(&ws[blockIdx.y], __float_as_uint(red[0]));
}

__device__ __forceinline__ int quant8(float x, float inv) {
    float r = rintf(x * inv);
    r = fminf(fmaxf(r, -128.f), 127.f);
    return (int)r;
}

__global__ __launch_bounds__(128)
void ibert_attn(const float* __restrict__ qg, const float* __restrict__ kg,
                const float* __restrict__ vg, const unsigned* __restrict__ ws,
                float* __restrict__ out) {
    __shared__ __align__(16) signed char   k8 [T_DIM * H_DIM];
    __shared__ __align__(16) signed char   vt8[H_DIM * T_DIM];
    __shared__ __align__(16) unsigned char P8 [4 * 16 * T_DIM];

    const int b    = blockIdx.y;
    const int tb   = blockIdx.x * 64;
    const int tid  = threadIdx.x;
    const int w    = tid >> 5;
    const int l    = tid & 31;
    const int half = l >> 4;
    const int m    = l & 15;

    const float qmax = __uint_as_float(ws[0]);
    const float kmax = __uint_as_float(ws[1]);
    const float vmax = __uint_as_float(ws[2]);
    const float q_sf = qmax * (1.0f / 127.0f);
    const float k_sf = kmax * (1.0f / 127.0f);
    const float v_sf = vmax * (1.0f / 127.0f);
    const float qinv = (qmax > 0.f) ? 127.0f / qmax : 0.f;
    const float kinv = (kmax > 0.f) ? 127.0f / kmax : 0.f;
    const float vinv = (vmax > 0.f) ? 127.0f / vmax : 0.f;
    const float sf   = q_sf * k_sf;

    const float4* kg4 = (const float4*)(kg + (size_t)b * T_DIM * H_DIM);
    const float4* vg4 = (const float4*)(vg + (size_t)b * T_DIM * H_DIM);
    for (int idx = tid; idx < T_DIM * H_DIM / 4; idx += 128) {
        int s  = idx >> 4;
        int c0 = (idx & 15) * 4;
        float4 kv = kg4[idx];
        int b0 = quant8(kv.x, kinv), b1 = quant8(kv.y, kinv);
        int b2 = quant8(kv.z, kinv), b3 = quant8(kv.w, kinv);
        *(int*)(k8 + s * H_DIM + c0) =
            (b0 & 255) | ((b1 & 255) << 8) | ((b2 & 255) << 16) | ((b3 & 255) << 24);
        float4 vv = vg4[idx];
        vt8[(c0 + 0) * T_DIM + s] = (signed char)quant8(vv.x, vinv);
        vt8[(c0 + 1) * T_DIM + s] = (signed char)quant8(vv.y, vinv);
        vt8[(c0 + 2) * T_DIM + s] = (signed char)quant8(vv.z, vinv);
        vt8[(c0 + 3) * T_DIM + s] = (signed char)quant8(vv.w, vinv);
    }
    __syncthreads();

    const int t0 = tb + 16 * w;
    v8i qa;
    const float* qrow = qg + ((size_t)(b * T_DIM + t0 + m)) * H_DIM;
#pragma unroll
    for (int j = 0; j < 8; ++j) {
        int c0 = 16 * (j >> 1) + 8 * half + 4 * (j & 1);
        float4 t4 = *(const float4*)(qrow + c0);
        qa[j] = (quant8(t4.x, qinv) & 255) | ((quant8(t4.y, qinv) & 255) << 8) |
                ((quant8(t4.z, qinv) & 255) << 16) | ((quant8(t4.w, qinv) & 255) << 24);
    }

    float xv[16][8];
    const float xmask = -1.0e9f / sf;
#pragma unroll
    for (int st = 0; st < 16; ++st) {
        int s0 = 16 * st;
        if (s0 <= t0 + 15) {
            v8i kb;
#pragma unroll
            for (int j = 0; j < 8; ++j) {
                int c0 = 16 * (j >> 1) + 8 * half + 4 * (j & 1);
                kb[j] = *(const int*)(k8 + (s0 + m) * H_DIM + c0);
            }
            v8i z = {0, 0, 0, 0, 0, 0, 0, 0};
            v8i acc = __builtin_amdgcn_wmma_i32_16x16x64_iu8(true, qa, true, kb, z, false, false);
#pragma unroll
            for (int g = 0; g < 8; ++g) {
                int trow = t0 + g + 8 * half;
                int scol = s0 + m;
                xv[st][g] = (scol <= trow) ? (float)acc[g] * 0.125f : xmask;
            }
        } else {
#pragma unroll
            for (int g = 0; g < 8; ++g) xv[st][g] = xmask;
        }
    }

    float rmax[8];
#pragma unroll
    for (int g = 0; g < 8; ++g) {
        float mm = -3.0e38f;
#pragma unroll
        for (int st = 0; st < 16; ++st) mm = fmaxf(mm, xv[st][g]);
        mm = fmaxf(mm, __shfl_xor(mm, 1, 32));
        mm = fmaxf(mm, __shfl_xor(mm, 2, 32));
        mm = fmaxf(mm, __shfl_xor(mm, 4, 32));
        mm = fmaxf(mm, __shfl_xor(mm, 8, 32));
        rmax[g] = mm;
    }

    const float x0i  = floorf(-0.6931f / sf);
    const float xmin = 30.0f * x0i;
    const float bint = floorf((float)(0.96963238 / 0.35815147) / sf);
    const float cint = floorf((float)(1.0 / 0.35815147) / (sf * sf));
    const float t16  = 32767.0f / (cint * 1073741824.0f);

    unsigned fac[8];
#pragma unroll
    for (int g = 0; g < 8; ++g) {
        int sum = 0;
#pragma unroll
        for (int st = 0; st < 16; ++st) {
            float x    = xv[st][g] - rmax[g];
            float xc   = fmaxf(x, xmin);
            float qf   = floorf(xc / x0i);
            float r    = xc - x0i * qf;
            float poly = (r + bint) * r + cint;
            float ei   = fmaxf(floorf(ldexpf(poly, 30 - (int)qf)), 0.0f);
            float e16f = fminf(rintf(ei * t16), 32767.0f);
            xv[st][g]  = e16f;
            sum += (int)e16f;
        }
        sum += __shfl_xor(sum, 1, 32);
        sum += __shfl_xor(sum, 2, 32);
        sum += __shfl_xor(sum, 4, 32);
        sum += __shfl_xor(sum, 8, 32);
        fac[g] = (unsigned)(4294967296ULL / (unsigned long long)(unsigned)sum);
    }

    unsigned char* Pw = P8 + w * 16 * T_DIM;
#pragma unroll
    for (int g = 0; g < 8; ++g) {
#pragma unroll
        for (int st = 0; st < 16; ++st) {
            unsigned e = (unsigned)xv[st][g];
            unsigned p = (unsigned)(((unsigned long long)e * fac[g]) >> 24);
            Pw[(g + 8 * half) * T_DIM + 16 * st + m] = (unsigned char)p;
        }
    }
    asm volatile("s_wait_dscnt 0" ::: "memory");

    v8i oacc[4];
#pragma unroll
    for (int ct = 0; ct < 4; ++ct) { v8i z = {0,0,0,0,0,0,0,0}; oacc[ct] = z; }
#pragma unroll
    for (int ks = 0; ks < 4; ++ks) {
        v8i pa;
#pragma unroll
        for (int j = 0; j < 8; ++j) {
            int c0 = 64 * ks + 16 * (j >> 1) + 8 * half + 4 * (j & 1);
            pa[j] = *(const int*)(Pw + m * T_DIM + c0);
        }
#pragma unroll
        for (int ct = 0; ct < 4; ++ct) {
            v8i vb;
#pragma unroll
            for (int j = 0; j < 8; ++j) {
                int s0 = 64 * ks + 16 * (j >> 1) + 8 * half + 4 * (j & 1);
                vb[j] = *(const int*)(vt8 + (16 * ct + m) * T_DIM + s0);
            }
            oacc[ct] = __builtin_amdgcn_wmma_i32_16x16x64_iu8(false, pa, true, vb,
                                                              oacc[ct], false, false);
        }
    }

    const float oscale = v_sf * (1.0f / 65536.0f);
#pragma unroll
    for (int ct = 0; ct < 4; ++ct) {
#pragma unroll
        for (int g = 0; g < 8; ++g) {
            int t = t0 + g + 8 * half;
            int c = 16 * ct + m;
            out[((size_t)(b * T_DIM + t)) * H_DIM + c] = (float)oacc[ct][g] * oscale;
        }
    }
}

extern "C" void kernel_launch(void* const* d_in, const int* in_sizes, int n_in,
                              void* d_out, int out_size, void* d_ws, size_t ws_size,
                              hipStream_t stream) {
    const float* q = (const float*)d_in[0];
    const float* k = (const float*)d_in[1];
    const float* v = (const float*)d_in[2];
    float* out = (float*)d_out;
    unsigned* ws = (unsigned*)d_ws;

    ibert_init<<<1, 1, 0, stream>>>(ws, out);
    absmax_kernel<<<dim3(256, 3), 256, 0, stream>>>(q, k, v, ws);
    ibert_attn<<<dim3(T_DIM / 64, B_DIM), 128, 0, stream>>>(q, k, v, ws, out);
}
